// PetalExpertUnit_44195213476101
// MI455X (gfx1250) — matrix operand order repaired and checked by reading, not compiled, not run on hardware
//
#include <hip/hip_runtime.h>
#include <hip/hip_bf16.h>

// ---------------------------------------------------------------------------
// PetalExpertUnit on gfx1250: 3x Mamba mixture.
// GEMMs -> v_wmma_f32_16x16x32_f16 ; selective scan -> 3-pass chunked
// associative scan (64 chunks x 64 steps) for wave32 WGP occupancy;
// final-proj weights staged into LDS via the Tensor Data Mover.
// ---------------------------------------------------------------------------

typedef __attribute__((ext_vector_type(16))) _Float16 v16h;
typedef __attribute__((ext_vector_type(8)))  float    v8f;
typedef __attribute__((ext_vector_type(4)))  unsigned int u32x4;
typedef __attribute__((ext_vector_type(8)))  int      i32x8;
typedef __attribute__((ext_vector_type(4)))  int      i32x4;

#define NSEQ 8      // 4 parts * B(=2)
#define LSEQ 4096   // D*H*W
#define CCH  256    // C_IN
#define DMC  64     // C/4
#define DIC  128    // 2*DMC
#define DSN  16     // state size
#define RTK  4      // dt rank
#define NDBL 36     // RTK + 2*DSN
#define NCHUNK 64   // scan chunks
#define CLEN  64    // LSEQ / NCHUNK

static __device__ __forceinline__ float silu_f(float x) {
    return x * (1.0f / (1.0f + __expf(-x)));
}
static __device__ __forceinline__ float softplus_f(float x) {
    return (x > 20.0f) ? x : log1pf(__expf(x));
}

// --- WMMA fragment loaders (wave32, contiguous k = 16*(lane/16) + s in A and B) ---------------------
static __device__ __forceinline__ v16h load_a_frag(const _Float16* src, int lda, int lane) {
    const int row = lane & 15;
    const int kb0 = (lane < 16) ? 0 : 16;
    const int kb1 = (lane < 16) ? 8 : 24;
    const _Float16* p = src + (size_t)row * lda;
    v16h a;
#pragma unroll
    for (int e = 0; e < 8; ++e) a[e] = p[kb0 + e];
#pragma unroll
    for (int e = 0; e < 8; ++e) a[8 + e] = p[kb1 + e];
    return a;
}
static __device__ __forceinline__ v16h load_b_frag(const _Float16* W, int ldk, int n0, int k0, int lane) {
    const int n  = n0 + (lane & 15);
    const int kb = k0 + ((lane < 16) ? 0 : 16);
    const _Float16* p = W + (size_t)n * ldk + kb;
    v16h b;
#pragma unroll
    for (int e = 0; e < 16; ++e) b[e] = p[e];
    return b;
}
static __device__ __forceinline__ v8f wmma_f16(v16h a, v16h b, v8f c) {
    return __builtin_amdgcn_wmma_f32_16x16x32_f16(false, a, false, b, (short)0, c, false, false);
}

// --- f32 -> f16 weight conversion ------------------------------------------
__global__ void cvt_f16_kernel(const float* __restrict__ s, _Float16* __restrict__ d, int n) {
    int i = blockIdx.x * 256 + threadIdx.x;
    if (i < n) d[i] = (_Float16)s[i];
}

// --- LN1: x (B,C,L) channel-strided -> parts (8,L,64) f32+f16 ---------------
__global__ void ln1_kernel(const float* __restrict__ x, const float* __restrict__ g,
                           const float* __restrict__ bt, float* __restrict__ parts_f32,
                           _Float16* __restrict__ parts_f16) {
    const int bl = blockIdx.x;          // 0..8191
    const int b  = bl >> 12;
    const int l  = bl & (LSEQ - 1);
    const int c  = threadIdx.x;         // 256
    float v = x[((size_t)b * CCH + c) * LSEQ + l];
    __shared__ float red[CCH];
    red[c] = v; __syncthreads();
#pragma unroll
    for (int s = 128; s > 0; s >>= 1) { if (c < s) red[c] += red[c + s]; __syncthreads(); }
    float mu = red[0] * (1.0f / CCH);
    __syncthreads();
    float dv = v - mu;
    red[c] = dv * dv; __syncthreads();
#pragma unroll
    for (int s = 128; s > 0; s >>= 1) { if (c < s) red[c] += red[c + s]; __syncthreads(); }
    float var = red[0] * (1.0f / CCH);
    float xn = dv * rsqrtf(var + 1e-5f) * g[c] + bt[c];
    const int p = c >> 6;
    const int seq = p * 2 + b;
    size_t idx = ((size_t)seq * LSEQ + l) * DMC + (c & 63);
    parts_f32[idx] = xn;
    parts_f16[idx] = (_Float16)xn;
}

// --- GEMM1: xz(32768x256) = parts_f16(32768x64) @ in_w^T(256x64) ------------
__global__ void gemm_inproj_kernel(const _Float16* __restrict__ A, const _Float16* __restrict__ W,
                                   float* __restrict__ C) {
    const int mt = blockIdx.x;          // 2048 M-tiles
    const int wave = threadIdx.x >> 5;  // 4 waves
    const int lane = threadIdx.x & 31;
    const int m0 = mt * 16;
    v8f acc[4] = {};
#pragma unroll
    for (int k0 = 0; k0 < 64; k0 += 32) {
        v16h a = load_a_frag(A + (size_t)m0 * 64 + k0, 64, lane);
#pragma unroll
        for (int j = 0; j < 4; ++j) {
            int n0 = (wave * 4 + j) * 16;
            v16h b = load_b_frag(W, 64, n0, k0, lane);
            acc[j] = wmma_f16(a, b, acc[j]);
        }
    }
    const int hi = (lane < 16) ? 0 : 8;
    const int col_l = lane & 15;
#pragma unroll
    for (int j = 0; j < 4; ++j) {
        int n0 = (wave * 4 + j) * 16;
#pragma unroll
        for (int v = 0; v < 8; ++v) {
            int m = m0 + v + hi;
            C[(size_t)m * CCH + n0 + col_l] = acc[j][v];
        }
    }
}

// --- Depthwise causal conv K=4 + SiLU: xi (cols 0..127 of xz) -> xc ---------
__global__ void conv_silu_kernel(const float* __restrict__ xz, const float* __restrict__ cw,
                                 const float* __restrict__ cb, float* __restrict__ xc) {
    const int l = blockIdx.x * 256 + threadIdx.x;
    const int d = blockIdx.y;
    const int n = blockIdx.z;
    const float* base = xz + ((size_t)n * LSEQ) * CCH + d;
    float acc = cb[d];
#pragma unroll
    for (int k = 0; k < 4; ++k) {
        int li = l + k - 3;
        float xv = (li >= 0) ? base[(size_t)li * CCH] : 0.0f;
        acc += cw[d * 4 + k] * xv;
    }
    float s = silu_f(acc);
    xc[((size_t)n * LSEQ + l) * DIC + d] = s;
}

// --- xdbl = xc @ xp_w^T (36), dt = softplus(xdbl[:4] @ dtp_w^T + b), B, C ---
__global__ void xdbl_dt_kernel(const float* __restrict__ xc, const float* __restrict__ xpw,
                               const float* __restrict__ dtw, const float* __restrict__ dtb,
                               float* __restrict__ dt, float* __restrict__ Bm,
                               float* __restrict__ Cm) {
    const size_t row = blockIdx.x;      // n*LSEQ + l, 32768 rows
    const int t = threadIdx.x;          // 128
    __shared__ float xrow[DIC];
    __shared__ float dlo[RTK];
    xrow[t] = xc[row * DIC + t];
    __syncthreads();
    if (t < NDBL) {
        const float* wr = xpw + t * DIC;
        float acc = 0.0f;
#pragma unroll 8
        for (int k = 0; k < DIC; ++k) acc += wr[k] * xrow[k];
        if (t < RTK)            dlo[t] = acc;
        else if (t < RTK + DSN) Bm[row * DSN + (t - RTK)] = acc;
        else                    Cm[row * DSN + (t - RTK - DSN)] = acc;
    }
    __syncthreads();
    float a = dtb[t];
#pragma unroll
    for (int r = 0; r < RTK; ++r) a += dlo[r] * dtw[t * RTK + r];
    dt[row * DIC + t] = softplus_f(a);
}

// --- Scan pass 1: per-chunk local state (h0=0) + per-state decay product ----
__global__ void scan_pass1_kernel(const float* __restrict__ dt, const float* __restrict__ xc,
                                  const float* __restrict__ Bm, const float* __restrict__ A_log,
                                  float* __restrict__ Pst, float* __restrict__ Hloc) {
    const int c = blockIdx.x & (NCHUNK - 1);
    const int n = blockIdx.x >> 6;
    const int d = threadIdx.x;          // 128
    float Ar[DSN], h[DSN], p[DSN];
#pragma unroll
    for (int s = 0; s < DSN; ++s) { Ar[s] = -__expf(A_log[d * DSN + s]); h[s] = 0.0f; p[s] = 1.0f; }
    const size_t rowbase = (size_t)n * LSEQ + c * CLEN;
    const float4* Bv = (const float4*)(Bm + rowbase * DSN);
    for (int l = 0; l < CLEN; ++l) {
        const size_t r = rowbase + l;
        float dtv = dt[r * DIC + d];
        float u   = xc[r * DIC + d];
        float Bl[DSN];
#pragma unroll
        for (int q = 0; q < 4; ++q) {
            float4 bq = Bv[l * 4 + q];
            Bl[q*4+0]=bq.x; Bl[q*4+1]=bq.y; Bl[q*4+2]=bq.z; Bl[q*4+3]=bq.w;
        }
#pragma unroll
        for (int s = 0; s < DSN; ++s) {
            float a = __expf(dtv * Ar[s]);
            h[s] = a * h[s] + dtv * Bl[s] * u;
            p[s] *= a;
        }
        if (l + 8 < CLEN) {
            __builtin_prefetch(&dt[(r + 8) * DIC + d], 0, 1);
            __builtin_prefetch(&Bv[(l + 8) * 4], 0, 1);
        }
    }
    const size_t o = (((size_t)n * NCHUNK + c) * DIC + d) * DSN;
    float4* Hp = (float4*)(Hloc + o);
    float4* Pp = (float4*)(Pst + o);
#pragma unroll
    for (int q = 0; q < 4; ++q) {
        Hp[q] = make_float4(h[q*4+0], h[q*4+1], h[q*4+2], h[q*4+3]);
        Pp[q] = make_float4(p[q*4+0], p[q*4+1], p[q*4+2], p[q*4+3]);
    }
}

// --- Scan pass 2: serial inter-chunk combine over 64 chunks -----------------
__global__ void scan_pass2_kernel(const float* __restrict__ Pst, const float* __restrict__ Hloc,
                                  float* __restrict__ Hin) {
    const int t = blockIdx.x * 256 + threadIdx.x;    // 16384 (n,d,s) lanes
    const int s = t & 15;
    const int d = (t >> 4) & 127;
    const int n = t >> 11;
    float hc = 0.0f;
    for (int c = 0; c < NCHUNK; ++c) {
        const size_t o = (((size_t)n * NCHUNK + c) * DIC + d) * DSN + s;
        Hin[o] = hc;
        hc = Pst[o] * hc + Hloc[o];
    }
}

// --- Scan pass 3: re-run each chunk from exact carry-in, emit gated output --
__global__ void scan_pass3_kernel(const float* __restrict__ dt, const float* __restrict__ xc,
                                  const float* __restrict__ Bm, const float* __restrict__ Cm,
                                  const float* __restrict__ A_log, const float* __restrict__ Dv,
                                  const float* __restrict__ xz, const float* __restrict__ Hin,
                                  _Float16* __restrict__ gated) {
    const int c = blockIdx.x & (NCHUNK - 1);
    const int n = blockIdx.x >> 6;
    const int d = threadIdx.x;          // 128
    float Ar[DSN], h[DSN];
#pragma unroll
    for (int s = 0; s < DSN; ++s) Ar[s] = -__expf(A_log[d * DSN + s]);
    const size_t o = (((size_t)n * NCHUNK + c) * DIC + d) * DSN;
    const float4* Hp = (const float4*)(Hin + o);
#pragma unroll
    for (int q = 0; q < 4; ++q) {
        float4 hq = Hp[q];
        h[q*4+0]=hq.x; h[q*4+1]=hq.y; h[q*4+2]=hq.z; h[q*4+3]=hq.w;
    }
    const float Dd = Dv[d];
    const size_t rowbase = (size_t)n * LSEQ + c * CLEN;
    const float4* Bv = (const float4*)(Bm + rowbase * DSN);
    const float4* Cv = (const float4*)(Cm + rowbase * DSN);
    for (int l = 0; l < CLEN; ++l) {
        const size_t r = rowbase + l;
        float dtv = dt[r * DIC + d];
        float u   = xc[r * DIC + d];
        float Bl[DSN], Cl[DSN];
#pragma unroll
        for (int q = 0; q < 4; ++q) {
            float4 bq = Bv[l * 4 + q];
            float4 cq = Cv[l * 4 + q];
            Bl[q*4+0]=bq.x; Bl[q*4+1]=bq.y; Bl[q*4+2]=bq.z; Bl[q*4+3]=bq.w;
            Cl[q*4+0]=cq.x; Cl[q*4+1]=cq.y; Cl[q*4+2]=cq.z; Cl[q*4+3]=cq.w;
        }
        float y = 0.0f;
#pragma unroll
        for (int s = 0; s < DSN; ++s) {
            h[s] = h[s] * __expf(dtv * Ar[s]) + dtv * Bl[s] * u;
            y += h[s] * Cl[s];
        }
        float yo = y + u * Dd;
        float zv = xz[r * CCH + DIC + d];
        gated[r * DIC + d] = (_Float16)(yo * silu_f(zv));
        if (l + 8 < CLEN) {
            __builtin_prefetch(&dt[(r + 8) * DIC + d], 0, 1);
            __builtin_prefetch(&Bv[(l + 8) * 4], 0, 1);
            __builtin_prefetch(&Cv[(l + 8) * 4], 0, 1);
        }
    }
}

// --- GEMM2: m = gated(32768x128) @ out_w^T(64x128); + skip; scatter to comb -
__global__ void gemm_outproj_kernel(const _Float16* __restrict__ A, const _Float16* __restrict__ W,
                                    const float* __restrict__ parts_f32,
                                    const float* __restrict__ skip_ptr,
                                    float* __restrict__ comb) {
    const int mt = blockIdx.x;          // 2048 M-tiles
    const int wave = threadIdx.x >> 5;  // 4 waves -> N=64
    const int lane = threadIdx.x & 31;
    const int m0 = mt * 16;
    const int n0 = wave * 16;
    v8f acc = {};
#pragma unroll
    for (int k0 = 0; k0 < DIC; k0 += 32) {
        v16h a = load_a_frag(A + (size_t)m0 * DIC + k0, DIC, lane);
        v16h b = load_b_frag(W, DIC, n0, k0, lane);
        acc = wmma_f16(a, b, acc);
    }
    const float skip = skip_ptr[0];
    const int hi = (lane < 16) ? 0 : 8;
    const int col = n0 + (lane & 15);
#pragma unroll
    for (int v = 0; v < 8; ++v) {
        int m = m0 + v + hi;
        int nseq = m >> 12;             // m / LSEQ
        int l = m & (LSEQ - 1);
        int p = nseq >> 1;
        int b = nseq & 1;
        float val = acc[v] + skip * parts_f32[(size_t)m * DMC + col];
        comb[(((size_t)b << 12) + l) * CCH + (p << 6) + col] = val;
    }
}

// --- LN2: comb (2,L,256) row-contiguous -> comb_f16 -------------------------
__global__ void ln2_kernel(const float* __restrict__ comb, const float* __restrict__ g,
                           const float* __restrict__ bt, _Float16* __restrict__ comb_f16) {
    const size_t bl = blockIdx.x;       // 0..8191
    const int c = threadIdx.x;
    float v = comb[bl * CCH + c];
    __shared__ float red[CCH];
    red[c] = v; __syncthreads();
#pragma unroll
    for (int s = 128; s > 0; s >>= 1) { if (c < s) red[c] += red[c + s]; __syncthreads(); }
    float mu = red[0] * (1.0f / CCH);
    __syncthreads();
    float dv = v - mu;
    red[c] = dv * dv; __syncthreads();
#pragma unroll
    for (int s = 128; s > 0; s >>= 1) { if (c < s) red[c] += red[c + s]; __syncthreads(); }
    float var = red[0] * (1.0f / CCH);
    comb_f16[bl * CCH + c] = (_Float16)(dv * rsqrtf(var + 1e-5f) * g[c] + bt[c]);
}

// --- GEMM3: out = comb_f16(8192x256) @ proj_w^T(256x256) + pb; masked accum -
// Weight slab (64 rows x 256 cols f16 = 32KB) staged into LDS via the TDM,
// reused across 4 M-tiles per block; B fragments served from LDS.
__global__ void gemm_proj_kernel(const _Float16* __restrict__ A, const _Float16* __restrict__ W,
                                 const float* __restrict__ pb, const int* __restrict__ dataset_id,
                                 int mix, float* __restrict__ out) {
    __shared__ _Float16 wlds[64 * CCH];            // 32 KB
    const int wave = threadIdx.x >> 5;
    const int lane = threadIdx.x & 31;
    const int nbase = blockIdx.y * 64;             // this block's 4 N-tiles

    if (wave == 0) {
        // Tensor DMA descriptor (ISA 8.3/8.4): 2D tile, 4-byte elements,
        // tile = 128 dwords x 64 rows, row stride 128 dwords.
        unsigned long long gaddr = (unsigned long long)(const void*)(W + (size_t)nbase * CCH);
        unsigned int laddr = (unsigned int)(unsigned long long)(const void*)wlds;
        u32x4 g0;
        g0[0] = 1u;                                            // count=1, user mode
        g0[1] = laddr;                                         // lds_addr (bytes)
        g0[2] = (unsigned int)gaddr;                           // global_addr[31:0]
        g0[3] = (unsigned int)((gaddr >> 32) & 0x1FFFFFFu) | (2u << 30);  // addr[56:32] | type=2
        i32x8 g1;
        g1[0] = (2 << 16);          // workgroup_mask=0, data_size=2 (4 bytes)
        g1[1] = (128 << 16);        // tensor_dim0[15:0]=128 (bits 63:48)
        g1[2] = (64 << 16);         // tensor_dim0 hi | tensor_dim1[15:0]=64
        g1[3] = (128 << 16);        // tensor_dim1 hi | tile_dim0=128
        g1[4] = 64;                 // tile_dim1=64, tile_dim2=0
        g1[5] = 128;                // tensor_dim0_stride[31:0]=128 dwords
        g1[6] = 0;                  // stride0 hi | tensor_dim1_stride lo
        g1[7] = 0;
        i32x4 g2 = {0, 0, 0, 0};
        i32x4 g3 = {0, 0, 0, 0};
        i32x8 g4 = {0, 0, 0, 0, 0, 0, 0, 0};       // extra group (clang-23 6-arg form)
        __builtin_amdgcn_tensor_load_to_lds(g0, g1, g2, g3, g4, 0);
        __builtin_amdgcn_s_wait_tensorcnt(0);
    }
    __syncthreads();

    const int hi = (lane < 16) ? 0 : 8;
    const int n0loc = wave * 16;                   // within the LDS slab
    const int col = nbase + n0loc + (lane & 15);
    const float bias = pb[col];
#pragma unroll
    for (int i = 0; i < 4; ++i) {
        const int m0 = (blockIdx.x * 4 + i) * 16;
        v8f acc = {};
#pragma unroll
        for (int k0 = 0; k0 < CCH; k0 += 32) {
            v16h a = load_a_frag(A + (size_t)m0 * CCH + k0, CCH, lane);
            v16h b = load_b_frag(wlds, CCH, n0loc, k0, lane);   // ds_load path
            acc = wmma_f16(a, b, acc);
        }
#pragma unroll
        for (int v = 0; v < 8; ++v) {
            int m = m0 + v + hi;
            int b = m >> 12;
            int l = m & (LSEQ - 1);
            float val = acc[v] + bias;
            float* dst = out + (((size_t)b << 8) + col) * LSEQ + l;  // (B, C_OUT, L)
            if (mix == 0) {
                *dst = val;
            } else {
                float sc = (dataset_id[b] == mix - 1) ? 1.0f : 0.0f;
                *dst += sc * val;
            }
        }
    }
}

// ---------------------------------------------------------------------------
extern "C" void kernel_launch(void* const* d_in, const int* in_sizes, int n_in,
                              void* d_out, int out_size, void* d_ws, size_t ws_size,
                              hipStream_t stream) {
    const float* x          = (const float*)d_in[0];
    const int*   dataset_id = (const int*)d_in[1];
    const float* norm_g     = (const float*)d_in[2];
    const float* norm_b     = (const float*)d_in[3];
    const float* in_proj_w  = (const float*)d_in[4];
    const float* conv_w     = (const float*)d_in[5];
    const float* conv_b     = (const float*)d_in[6];
    const float* x_proj_w   = (const float*)d_in[7];
    const float* dt_proj_w  = (const float*)d_in[8];
    const float* dt_proj_b  = (const float*)d_in[9];
    const float* A_log      = (const float*)d_in[10];
    const float* D_ssm      = (const float*)d_in[11];
    const float* out_proj_w = (const float*)d_in[12];
    const float* proj_w     = (const float*)d_in[13];
    const float* proj_b     = (const float*)d_in[14];
    const float* skip_scale = (const float*)d_in[15];
    float* out = (float*)d_out;

    // Workspace carve (reused across mixes)
    char* ws = (char*)d_ws;
    size_t off = 0;
    auto carve = [&](size_t bytes) { void* p = ws + off; off += (bytes + 255) & ~(size_t)255; return p; };
    float*    parts_f32 = (float*)   carve((size_t)NSEQ * LSEQ * DMC * 4);   // 8 MB
    _Float16* parts_f16 = (_Float16*)carve((size_t)NSEQ * LSEQ * DMC * 2);   // 4 MB
    float*    xz        = (float*)   carve((size_t)NSEQ * LSEQ * CCH * 4);   // 32 MB
    float*    xc        = (float*)   carve((size_t)NSEQ * LSEQ * DIC * 4);   // 16 MB
    float*    dtb       = (float*)   carve((size_t)NSEQ * LSEQ * DIC * 4);   // 16 MB
    float*    Bm        = (float*)   carve((size_t)NSEQ * LSEQ * DSN * 4);   // 2 MB
    float*    Cm        = (float*)   carve((size_t)NSEQ * LSEQ * DSN * 4);   // 2 MB
    _Float16* gated     = (_Float16*)carve((size_t)NSEQ * LSEQ * DIC * 2);   // 8 MB
    float*    comb      = (float*)   carve((size_t)2 * LSEQ * CCH * 4);      // 8 MB
    _Float16* comb_f16  = (_Float16*)carve((size_t)2 * LSEQ * CCH * 2);      // 4 MB
    float*    Pst       = (float*)   carve((size_t)NSEQ * NCHUNK * DIC * DSN * 4);  // 4 MB
    float*    Hloc      = (float*)   carve((size_t)NSEQ * NCHUNK * DIC * DSN * 4);  // 4 MB
    float*    Hin       = (float*)   carve((size_t)NSEQ * NCHUNK * DIC * DSN * 4);  // 4 MB
    _Float16* w_in_h    = (_Float16*)carve((size_t)CCH * DMC * 2);
    _Float16* w_out_h   = (_Float16*)carve((size_t)DMC * DIC * 2);
    _Float16* w_proj_h  = (_Float16*)carve((size_t)CCH * CCH * 2);

    for (int mix = 0; mix < 3; ++mix) {
        const float* g    = norm_g     + mix * CCH;
        const float* bt   = norm_b     + mix * CCH;
        const float* inw  = in_proj_w  + mix * CCH * DMC;
        const float* cw   = conv_w     + mix * DIC * 4;
        const float* cb   = conv_b     + mix * DIC;
        const float* xpw  = x_proj_w   + mix * NDBL * DIC;
        const float* dtw  = dt_proj_w  + mix * DIC * RTK;
        const float* dtbp = dt_proj_b  + mix * DIC;
        const float* Alg  = A_log      + mix * DIC * DSN;
        const float* Dv   = D_ssm      + mix * DIC;
        const float* outw = out_proj_w + mix * DMC * DIC;
        const float* pw   = proj_w     + mix * CCH * CCH;
        const float* pb   = proj_b     + mix * CCH;
        const float* skip = skip_scale + mix;

        // Weight conversion to f16
        cvt_f16_kernel<<<(CCH * DMC + 255) / 256, 256, 0, stream>>>(inw,  w_in_h,   CCH * DMC);
        cvt_f16_kernel<<<(DMC * DIC + 255) / 256, 256, 0, stream>>>(outw, w_out_h,  DMC * DIC);
        cvt_f16_kernel<<<(CCH * CCH + 255) / 256, 256, 0, stream>>>(pw,   w_proj_h, CCH * CCH);

        // LN1 + split into parts
        ln1_kernel<<<2 * LSEQ, CCH, 0, stream>>>(x, g, bt, parts_f32, parts_f16);

        // in_proj GEMM (WMMA): xz = parts @ in_w^T
        gemm_inproj_kernel<<<(NSEQ * LSEQ) / 16, 128, 0, stream>>>(parts_f16, w_in_h, xz);

        // depthwise causal conv + SiLU
        conv_silu_kernel<<<dim3(LSEQ / 256, DIC, NSEQ), 256, 0, stream>>>(xz, cw, cb, xc);

        // xdbl / dt / B / C
        xdbl_dt_kernel<<<NSEQ * LSEQ, DIC, 0, stream>>>(xc, xpw, dtw, dtbp, dtb, Bm, Cm);

        // chunked selective scan (3 passes) + z-gating
        scan_pass1_kernel<<<NSEQ * NCHUNK, DIC, 0, stream>>>(dtb, xc, Bm, Alg, Pst, Hloc);
        scan_pass2_kernel<<<(NSEQ * DIC * DSN) / 256, 256, 0, stream>>>(Pst, Hloc, Hin);
        scan_pass3_kernel<<<NSEQ * NCHUNK, DIC, 0, stream>>>(dtb, xc, Bm, Cm, Alg, Dv, xz, Hin, gated);

        // out_proj GEMM (WMMA) + skip, scatter into combined (B,L,C)
        gemm_outproj_kernel<<<(NSEQ * LSEQ) / 16, 128, 0, stream>>>(gated, w_out_h, parts_f32, skip, comb);

        // LN2
        ln2_kernel<<<2 * LSEQ, CCH, 0, stream>>>(comb, g, bt, comb_f16);

        // final proj GEMM (WMMA + TDM weight staging) + bias + masked accumulate
        gemm_proj_kernel<<<dim3((2 * LSEQ) / 64, 4), 128, 0, stream>>>(comb_f16, w_proj_h, pb,
                                                                       dataset_id, mix, out);
    }
    (void)in_sizes; (void)n_in; (void)out_size; (void)ws_size;
}
